// Posterior3D_spec_80831284511070
// MI455X (gfx1250) — matrix operand order repaired and checked by reading, not compiled, not run on hardware
//
#include <hip/hip_runtime.h>
#include <hip/hip_bf16.h>
#include <cstdint>

#define NB     8
#define NGRID  4096
#define NOBS   4096
#define NPER   32
#define NLAYER 32
#define NCAX   50
#define NCOL   256   // NB * NPER fused columns

typedef __bf16 bf16_t;
typedef __attribute__((ext_vector_type(4)))  __bf16 v4bf;
typedef __attribute__((ext_vector_type(8)))  __bf16 v8bf;
typedef __attribute__((ext_vector_type(16))) __bf16 v16bf;
typedef __attribute__((ext_vector_type(8)))  float  v8f;

#define CAT16(lo, hi) __builtin_shufflevector((lo), (hi), 0,1,2,3,4,5,6,7,8,9,10,11,12,13,14,15)

// ---------------- workspace layout (bytes) ----------------
// emax     @ 0         : NOBS*NPER*4      = 524288
// logZ     @ 524288    : 524288
// S_T      @ 1048576   : NCOL*NGRID*2     = 2097152   (bf16, [n][g] K-contiguous)
// dspT     @ 3145728   : NCOL*NOBS*2      = 2097152   (bf16, [n][o])
// lognumT  @ 5242880   : NCOL*NOBS*4      = 4194304   (f32,  [n][o])
// dcT      @ 9437184   : NCOL*NGRID*4     = 4194304   (f32,  [n][g])
// partial  @ 13631488  : NB*512*4         = 16384
// scale    @ 13647872  : NB*4
// ATbf     @ 16777216  : NGRID*NOBS*2     = 33554432  (bf16, [g][o] K-contiguous)
// total ~48 MB

// ---------------------------------------------------------------------------
// Kernel 0: A^T (bf16) builder: AT[g][o] = bf16(A[o][g]); 64x64 LDS tiles,
// coalesced float4 reads and b64 writes on both sides. Warms A into L2.
// ---------------------------------------------------------------------------
__global__ __launch_bounds__(256) void transpose_kernel(
    const float* __restrict__ A, bf16_t* __restrict__ AT) {
    __shared__ __align__(16) bf16_t Tls[64 * 68];   // pitch 68 halves = 136B (8B-aligned rows)
    const int t = threadIdx.x;
    const int o0 = blockIdx.x * 64;
    const int g0 = blockIdx.y * 64;
    #pragma unroll
    for (int j = 0; j < 4; ++j) {
        int r  = (t >> 4) + j * 16;       // o within tile
        int c4 = (t & 15) * 4;            // g within tile
        float4 v = *(const float4*)(&A[(size_t)(o0 + r) * NGRID + g0 + c4]);
        v4bf w; w[0] = (bf16_t)v.x; w[1] = (bf16_t)v.y; w[2] = (bf16_t)v.z; w[3] = (bf16_t)v.w;
        *(v4bf*)(&Tls[r * 68 + c4]) = w;
    }
    __syncthreads();
    #pragma unroll
    for (int j = 0; j < 4; ++j) {
        int gg  = (t >> 4) + j * 16;      // g within tile
        int oo4 = (t & 15) * 4;           // o within tile
        v4bf w;
        w[0] = Tls[(oo4 + 0) * 68 + gg];
        w[1] = Tls[(oo4 + 1) * 68 + gg];
        w[2] = Tls[(oo4 + 2) * 68 + gg];
        w[3] = Tls[(oo4 + 3) * 68 + gg];
        *(v4bf*)(&AT[(size_t)(g0 + gg) * NOBS + o0 + oo4]) = w;
    }
}

// ---------------------------------------------------------------------------
// Kernel 1: per-(o,p) emax + trapezoidal logZ over the 50-point c-axis.
// ---------------------------------------------------------------------------
__global__ void logz_kernel(const float* __restrict__ E_obs,
                            const float* __restrict__ c_axis,
                            float* __restrict__ emax,
                            float* __restrict__ logZ) {
    int idx = blockIdx.x * blockDim.x + threadIdx.x;   // 0 .. NOBS*NPER-1
    int o = idx >> 5, p = idx & 31;
    const float* E  = E_obs + (size_t)o * (NCAX * NPER) + p;
    const float* cx = c_axis + o * NCAX;
    float m = -1e30f;
    for (int c = 0; c < NCAX; ++c) m = fmaxf(m, E[c * NPER]);
    float prev = expf(-(m - E[0]));
    float z = 0.0f;
    for (int c = 1; c < NCAX; ++c) {
        float cur = expf(-(m - E[c * NPER]));
        z += 0.5f * (prev + cur) * (cx[c] - cx[c - 1]);
        prev = cur;
    }
    emax[idx] = m;
    logZ[idx] = logf(z + 1e-30f);
}

// ---------------------------------------------------------------------------
// Kernel 2: pack s_true = 1/c_true into S_T [n][g] bf16 (K-contiguous rows).
// Writes coalesced; 4 MB gather on c_true is L2-resident after first touch.
// ---------------------------------------------------------------------------
__global__ void pack_s_kernel(const float* __restrict__ c_true,
                              bf16_t* __restrict__ ST) {
    int idx = blockIdx.x * blockDim.x + threadIdx.x;   // NCOL*NGRID
    int n = idx >> 12, g = idx & 4095;
    int b = n >> 5,   p = n & 31;
    float c = c_true[((size_t)b * NGRID + g) * NPER + p];
    ST[idx] = (bf16_t)(1.0f / c);
}

// ---------------------------------------------------------------------------
// Shared GEMM core pieces: block tile 64(M) x 64(N), K-step 64, 8 waves
// (4 along M x 2 along N, each wave owns two 16x16 f32 accumulators).
// LDS pitch 72 halves = 144 B (16B multiple -> all fragment loads are
// ds_load_b128; bank stride 36 -> 16 lanes on 16 distinct banks).
// ---------------------------------------------------------------------------

// Kernel 3: GEMM1  s_pred = A @ S  -> fused likelihood epilogue.
__global__ __launch_bounds__(256) void gemm1_kernel(
    const float*  __restrict__ A,      // f32 [o][g]
    const bf16_t* __restrict__ ST,     // bf16 [n][g]
    const float*  __restrict__ E_obs,
    const float*  __restrict__ c_axis,
    const float*  __restrict__ emax,
    const float*  __restrict__ logZ,
    float*  __restrict__ lognumT,      // f32 [n][o]
    bf16_t* __restrict__ dspT) {       // bf16 [n][o]
    __shared__ __align__(16) bf16_t Als[64 * 72];   // [m][k]
    __shared__ __align__(16) bf16_t Bls[64 * 72];   // [n][k]

    const int t = threadIdx.x;
    const int o0 = blockIdx.x * 64;
    const int n0 = blockIdx.y * 64;
    const int wid = t >> 5, lane = t & 31;
    const int m_off = (wid & 3) * 16;
    const int n_off = (wid >> 2) * 32;
    const int rowA = m_off + (lane & 15);
    const int colB = lane & 15;
    const int kb   = (lane < 16) ? 0 : 16;   // A frag K split (contiguous, matches B)
    const int kb2  = (lane < 16) ? 0 : 16;   // B frag K split

    v8f acc0 = {}; v8f acc1 = {};

    for (int k0 = 0; k0 < NGRID; k0 += 64) {
        // stage A tile (64x64 f32 -> bf16): float4 loads, b64 LDS stores
        #pragma unroll
        for (int j = 0; j < 4; ++j) {
            int r  = (t >> 4) + j * 16;
            int c4 = (t & 15) * 4;
            float4 v = *(const float4*)(&A[(size_t)(o0 + r) * NGRID + k0 + c4]);
            v4bf w; w[0] = (bf16_t)v.x; w[1] = (bf16_t)v.y; w[2] = (bf16_t)v.z; w[3] = (bf16_t)v.w;
            *(v4bf*)(&Als[r * 72 + c4]) = w;
        }
        // stage B tile (64 n x 64 k bf16): 16B loads/stores
        #pragma unroll
        for (int j = 0; j < 2; ++j) {
            int n  = (t >> 3) + j * 32;
            int kk = (t & 7) * 8;
            *(v8bf*)(&Bls[n * 72 + kk]) = *(const v8bf*)(&ST[(size_t)(n0 + n) * NGRID + k0 + kk]);
        }
        __syncthreads();

        #pragma unroll
        for (int h = 0; h < 2; ++h) {
            const bf16_t* ap = &Als[rowA * 72 + h * 32 + kb];
            v8bf a0 = *(const v8bf*)(ap);
            v8bf a1 = *(const v8bf*)(ap + 8);
            v16bf a = CAT16(a0, a1);
            const bf16_t* bp0 = &Bls[(n_off + colB) * 72 + h * 32 + kb2];
            v16bf b0 = CAT16(*(const v8bf*)(bp0), *(const v8bf*)(bp0 + 8));
            const bf16_t* bp1 = &Bls[(n_off + 16 + colB) * 72 + h * 32 + kb2];
            v16bf b1 = CAT16(*(const v8bf*)(bp1), *(const v8bf*)(bp1 + 8));
            acc0 = __builtin_amdgcn_wmma_f32_16x16x32_bf16(false, a, false, b0, (short)0, acc0, false, false);
            acc1 = __builtin_amdgcn_wmma_f32_16x16x32_bf16(false, a, false, b1, (short)0, acc1, false, false);
        }
        __syncthreads();
    }

    // fused epilogue: per lane each half has fixed n, 8 consecutive o
    const int obase = o0 + m_off + ((lane >> 4) << 3);
    #pragma unroll
    for (int half = 0; half < 2; ++half) {
        v8f acc = half ? acc1 : acc0;
        const int ncol = n0 + n_off + half * 16 + colB;
        const int p = ncol & 31;
        v8f ln; v8bf dv;
        #pragma unroll
        for (int i = 0; i < 8; ++i) {
            int o = obase + i;
            float sp = acc[i];
            float cp = 1.0f / sp;
            const float* cx = c_axis + o * NCAX;
            int idx = 0;                       // searchsorted 'left' = #(cx < cp)
            for (int c = 0; c < NCAX; ++c) idx += (cx[c] < cp) ? 1 : 0;
            idx = min(max(idx, 1), NCAX - 1);
            float x0 = cx[idx - 1], x1 = cx[idx];
            const float* E = E_obs + (size_t)o * (NCAX * NPER) + p;
            float y0 = E[(idx - 1) * NPER], y1 = E[idx * NPER];
            float inv_dx = 1.0f / (x1 - x0 + 1e-12f);
            float tt    = (cp - x0) * inv_dx;
            float epred = y0 + tt * (y1 - y0);
            float dedc  = (y1 - y0) * inv_dx;
            int op = o * NPER + p;
            ln[i] = -(emax[op] - epred) - logZ[op];
            dv[i] = (bf16_t)(dedc * (-1.0f / (sp * sp)));
        }
        *(v8f*) (&lognumT[(size_t)ncol * NOBS + obase]) = ln;
        *(v8bf*)(&dspT  [(size_t)ncol * NOBS + obase]) = dv;
    }
}

// ---------------------------------------------------------------------------
// Kernel 4: log_like[b] = (1/NPER) * sum_{p,o} lognumT[b*32+p][o] (fixed tree)
// ---------------------------------------------------------------------------
__global__ void loglike_kernel(const float* __restrict__ lognumT,
                               float* __restrict__ out) {
    __shared__ float red[256];
    int b = blockIdx.x, t = threadIdx.x;
    float s = 0.0f;
    for (int i = t; i < NPER * NOBS; i += 256) {
        int p = i >> 12, o = i & 4095;
        s += lognumT[(size_t)(b * NPER + p) * NOBS + o];
    }
    red[t] = s; __syncthreads();
    for (int w = 128; w > 0; w >>= 1) { if (t < w) red[t] += red[t + w]; __syncthreads(); }
    if (t == 0) out[b] = red[0] * (1.0f / NPER);
}

// ---------------------------------------------------------------------------
// Kernel 5: GEMM2  dlogp_ds_true = A^T @ dsp, fused * (-1/c_true^2) -> dcT[n][g]
// Identical data path to GEMM1: AT is pre-built bf16 [g][o] K-contiguous.
// ---------------------------------------------------------------------------
__global__ __launch_bounds__(256) void gemm2_kernel(
    const bf16_t* __restrict__ AT,     // bf16 [g][o]
    const bf16_t* __restrict__ dspT,   // bf16 [n][o]
    const float*  __restrict__ c_true,
    float* __restrict__ dcT) {         // f32 [n][g]
    __shared__ __align__(16) bf16_t Als[64 * 72];   // [g][k=o]
    __shared__ __align__(16) bf16_t Bls[64 * 72];   // [n][k=o]

    const int t = threadIdx.x;
    const int g0 = blockIdx.x * 64;
    const int n0 = blockIdx.y * 64;
    const int wid = t >> 5, lane = t & 31;
    const int m_off = (wid & 3) * 16;
    const int n_off = (wid >> 2) * 32;
    const int rowA = m_off + (lane & 15);
    const int colB = lane & 31 & 15;
    const int kb   = (lane < 16) ? 0 : 16;
    const int kb2  = (lane < 16) ? 0 : 16;

    v8f acc0 = {}; v8f acc1 = {};

    for (int k0 = 0; k0 < NOBS; k0 += 64) {
        #pragma unroll
        for (int j = 0; j < 2; ++j) {
            int g  = (t >> 3) + j * 32;
            int kk = (t & 7) * 8;
            *(v8bf*)(&Als[g * 72 + kk]) = *(const v8bf*)(&AT[(size_t)(g0 + g) * NOBS + k0 + kk]);
        }
        #pragma unroll
        for (int j = 0; j < 2; ++j) {
            int n  = (t >> 3) + j * 32;
            int kk = (t & 7) * 8;
            *(v8bf*)(&Bls[n * 72 + kk]) = *(const v8bf*)(&dspT[(size_t)(n0 + n) * NOBS + k0 + kk]);
        }
        __syncthreads();

        #pragma unroll
        for (int h = 0; h < 2; ++h) {
            const bf16_t* ap = &Als[rowA * 72 + h * 32 + kb];
            v16bf a = CAT16(*(const v8bf*)(ap), *(const v8bf*)(ap + 8));
            const bf16_t* bp0 = &Bls[(n_off + colB) * 72 + h * 32 + kb2];
            v16bf b0 = CAT16(*(const v8bf*)(bp0), *(const v8bf*)(bp0 + 8));
            const bf16_t* bp1 = &Bls[(n_off + 16 + colB) * 72 + h * 32 + kb2];
            v16bf b1 = CAT16(*(const v8bf*)(bp1), *(const v8bf*)(bp1 + 8));
            acc0 = __builtin_amdgcn_wmma_f32_16x16x32_bf16(false, a, false, b0, (short)0, acc0, false, false);
            acc1 = __builtin_amdgcn_wmma_f32_16x16x32_bf16(false, a, false, b1, (short)0, acc1, false, false);
        }
        __syncthreads();
    }

    const int gbase = g0 + m_off + ((lane >> 4) << 3);
    #pragma unroll
    for (int half = 0; half < 2; ++half) {
        v8f acc = half ? acc1 : acc0;
        const int ncol = n0 + n_off + half * 16 + colB;
        const int b = ncol >> 5, p = ncol & 31;
        v8f dcv;
        #pragma unroll
        for (int i = 0; i < 8; ++i) {
            float ct = c_true[((size_t)b * NGRID + gbase + i) * NPER + p];
            dcv[i] = acc[i] * (-1.0f / (ct * ct));
        }
        *(v8f*)(&dcT[(size_t)ncol * NGRID + gbase]) = dcv;
    }
}

// ---------------------------------------------------------------------------
// Kernel 6: grad[b,g,l] = sum_p J[b,g,p,l] * dc[b,g,p]; one wave per (b,g).
// Streams the 128 MB J fully coalesced; dcT gather is L2-resident (4 MB).
// ---------------------------------------------------------------------------
__global__ __launch_bounds__(256) void grad_kernel(
    const float* __restrict__ J,
    const float* __restrict__ dcT,     // f32 [n][g]
    float* __restrict__ gout,          // d_out + 8
    float* __restrict__ partial) {
    __shared__ float red[256];
    const int t = threadIdx.x, wid = t >> 5, lane = t & 31;
    const int pair = blockIdx.x * 8 + wid;           // = b*NGRID + g ; block never straddles b
    const int b = pair >> 12, g = pair & 4095;
    const float* Jp = J + (size_t)pair * (NPER * NLAYER);
    float dval = dcT[(size_t)(b * NPER + lane) * NGRID + g];
    float acc = 0.0f;
    #pragma unroll
    for (int p = 0; p < NPER; ++p) {
        float dp = __shfl(dval, p, 32);
        acc += Jp[p * NLAYER + lane] * dp;
    }
    gout[(size_t)pair * NLAYER + lane] = acc;
    red[t] = acc * acc; __syncthreads();
    for (int w = 128; w > 0; w >>= 1) { if (t < w) red[t] += red[t + w]; __syncthreads(); }
    if (t == 0) partial[blockIdx.x] = red[0];        // blockIdx = b*512 + block_in_b
}

// ---------------------------------------------------------------------------
// Kernel 7: per-batch norm + clip factor
// ---------------------------------------------------------------------------
__global__ void norm_kernel(const float* __restrict__ partial,
                            float* __restrict__ scale) {
    __shared__ float red[256];
    int b = blockIdx.x, t = threadIdx.x;
    red[t] = partial[b * 512 + t] + partial[b * 512 + 256 + t];
    __syncthreads();
    for (int w = 128; w > 0; w >>= 1) { if (t < w) red[t] += red[t + w]; __syncthreads(); }
    if (t == 0) {
        float n = sqrtf(red[0]);
        scale[b] = (n > 1000.0f) ? 1000.0f / n : 1.0f;
    }
}

// ---------------------------------------------------------------------------
// Kernel 8: apply clip scale in place
// ---------------------------------------------------------------------------
__global__ void scale_kernel(float* __restrict__ gout,
                             const float* __restrict__ scale) {
    int i = blockIdx.x * blockDim.x + threadIdx.x;   // NB*NGRID*NLAYER
    int b = i >> 17;                                 // NGRID*NLAYER = 2^17
    gout[i] *= scale[b];
}

// ---------------------------------------------------------------------------
extern "C" void kernel_launch(void* const* d_in, const int* in_sizes, int n_in,
                              void* d_out, int out_size, void* d_ws, size_t ws_size,
                              hipStream_t stream) {
    const float* A      = (const float*)d_in[0];
    const float* c_true = (const float*)d_in[1];
    const float* J      = (const float*)d_in[2];
    const float* E_obs  = (const float*)d_in[3];
    const float* c_axis = (const float*)d_in[4];
    float* out = (float*)d_out;

    char* ws = (char*)d_ws;
    float*  emax    = (float*)(ws);
    float*  logZ    = (float*)(ws + 524288);
    bf16_t* ST      = (bf16_t*)(ws + 1048576);
    bf16_t* dspT    = (bf16_t*)(ws + 3145728);
    float*  lognumT = (float*)(ws + 5242880);
    float*  dcT     = (float*)(ws + 9437184);
    float*  partial = (float*)(ws + 13631488);
    float*  scalev  = (float*)(ws + 13647872);
    bf16_t* AT      = (bf16_t*)(ws + 16777216);

    transpose_kernel<<<dim3(NOBS / 64, NGRID / 64), 256, 0, stream>>>(A, AT);
    logz_kernel     <<<NOBS * NPER / 256, 256, 0, stream>>>(E_obs, c_axis, emax, logZ);
    pack_s_kernel   <<<NCOL * NGRID / 256, 256, 0, stream>>>(c_true, ST);
    gemm1_kernel    <<<dim3(NOBS / 64, NCOL / 64), 256, 0, stream>>>(A, ST, E_obs, c_axis,
                                                                     emax, logZ, lognumT, dspT);
    loglike_kernel  <<<NB, 256, 0, stream>>>(lognumT, out);
    gemm2_kernel    <<<dim3(NGRID / 64, NCOL / 64), 256, 0, stream>>>(AT, dspT, c_true, dcT);
    grad_kernel     <<<NB * NGRID / 8, 256, 0, stream>>>(J, dcT, out + 8, partial);
    norm_kernel     <<<NB, 256, 0, stream>>>(partial, scalev);
    scale_kernel    <<<NB * NGRID * NLAYER / 256, 256, 0, stream>>>(out + 8, scalev);
}
